// Attention_59141699666413
// MI455X (gfx1250) — matrix operand order repaired and checked by reading, not compiled, not run on hardware
//
#include <hip/hip_runtime.h>

// ---------------------------------------------------------------------------
// Attention: out = softmax(mask ? (XqWq')(XkWk')^T/sqrt(128) : -1e9) (XvWv')
// B=8, S=2048, M=1024, DK=128.  All GEMMs via v_wmma_f32_16x16x32_f16.
// ---------------------------------------------------------------------------

typedef __attribute__((ext_vector_type(16))) _Float16 v16h;
typedef __attribute__((ext_vector_type(8)))  _Float16 v8h;
typedef __attribute__((ext_vector_type(8)))  float    v8f;

#define WMMA_F16(a, b, c) \
  __builtin_amdgcn_wmma_f32_16x16x32_f16(false, (a), false, (b), (short)0, (c), false, false)

#define BATCH  8
#define SEQ    2048
#define MDIM   1024
#define DK     128

// ---- fragment loaders ------------------------------------------------------
// A (16x32 f16, MxK): lane<16 -> row M=lane, K = k0..k0+15
//                     lane>=16 -> row M=lane-16, K = k0+16..k0+31
__device__ __forceinline__ v16h load_a_f32(const float* __restrict__ src, int ld,
                                           int row0, int k0, int lane) {
  const int hi = (lane >> 4) & 1, lr = lane & 15;
  const float* p = src + (size_t)(row0 + lr) * ld + k0;
  const float4 x0 = *reinterpret_cast<const float4*>(p + hi * 16);
  const float4 x1 = *reinterpret_cast<const float4*>(p + hi * 16 + 4);
  const float4 x2 = *reinterpret_cast<const float4*>(p + hi * 16 + 8);
  const float4 x3 = *reinterpret_cast<const float4*>(p + hi * 16 + 12);
  v16h a;
  a[0]  = (_Float16)x0.x; a[1]  = (_Float16)x0.y; a[2]  = (_Float16)x0.z; a[3]  = (_Float16)x0.w;
  a[4]  = (_Float16)x1.x; a[5]  = (_Float16)x1.y; a[6]  = (_Float16)x1.z; a[7]  = (_Float16)x1.w;
  a[8]  = (_Float16)x2.x; a[9]  = (_Float16)x2.y; a[10] = (_Float16)x2.z; a[11] = (_Float16)x2.w;
  a[12] = (_Float16)x3.x; a[13] = (_Float16)x3.y; a[14] = (_Float16)x3.z; a[15] = (_Float16)x3.w;
  return a;
}

// B (32x16 f16, KxN) from a [N][K] row-major ("transposed") f32 source:
// lane -> column N = n0 + (lane&15); K = k0 + (lane<16 ? 0..15 : 16..31)
__device__ __forceinline__ v16h load_b_f32t(const float* __restrict__ srcT, int ld,
                                            int n0, int k0, int lane) {
  const int hi = (lane >> 4) & 1, lr = lane & 15;
  const float* p = srcT + (size_t)(n0 + lr) * ld + k0 + hi * 16;
  const float4 x0 = *reinterpret_cast<const float4*>(p + 0);
  const float4 x1 = *reinterpret_cast<const float4*>(p + 4);
  const float4 x2 = *reinterpret_cast<const float4*>(p + 8);
  const float4 x3 = *reinterpret_cast<const float4*>(p + 12);
  v16h b;
  b[0]  = (_Float16)x0.x; b[1]  = (_Float16)x0.y; b[2]  = (_Float16)x0.z; b[3]  = (_Float16)x0.w;
  b[4]  = (_Float16)x1.x; b[5]  = (_Float16)x1.y; b[6]  = (_Float16)x1.z; b[7]  = (_Float16)x1.w;
  b[8]  = (_Float16)x2.x; b[9]  = (_Float16)x2.y; b[10] = (_Float16)x2.z; b[11] = (_Float16)x2.w;
  b[12] = (_Float16)x3.x; b[13] = (_Float16)x3.y; b[14] = (_Float16)x3.z; b[15] = (_Float16)x3.w;
  return b;
}

__device__ __forceinline__ v16h load_a_f16(const _Float16* __restrict__ src, int ld,
                                           int row0, int k0, int lane) {
  const int hi = (lane >> 4) & 1, lr = lane & 15;
  const _Float16* p = src + (size_t)(row0 + lr) * ld + k0;
  v8h lo  = *reinterpret_cast<const v8h*>(p + hi * 16);
  v8h hi8 = *reinterpret_cast<const v8h*>(p + hi * 16 + 8);
  return __builtin_shufflevector(lo, hi8, 0, 1, 2, 3, 4, 5, 6, 7,
                                 8, 9, 10, 11, 12, 13, 14, 15);
}

__device__ __forceinline__ v16h load_b_f16(const _Float16* __restrict__ srcT, int ld,
                                           int n0, int k0, int lane) {
  const int hi = (lane >> 4) & 1, lr = lane & 15;
  return *reinterpret_cast<const v16h*>(srcT + (size_t)(n0 + lr) * ld + k0 + hi * 16);
}

__device__ __forceinline__ void store8_f16(_Float16* dst, v8f c) {
  v8h h;
#pragma unroll
  for (int e = 0; e < 8; ++e) h[e] = (_Float16)c[e];
  *reinterpret_cast<v8h*>(dst) = h;
}

__device__ __forceinline__ float redmax16(float v) {
  v = fmaxf(v, __shfl_xor(v, 1, 32));
  v = fmaxf(v, __shfl_xor(v, 2, 32));
  v = fmaxf(v, __shfl_xor(v, 4, 32));
  v = fmaxf(v, __shfl_xor(v, 8, 32));
  return v;
}
__device__ __forceinline__ float redsum16(float v) {
  v += __shfl_xor(v, 1, 32);
  v += __shfl_xor(v, 2, 32);
  v += __shfl_xor(v, 4, 32);
  v += __shfl_xor(v, 8, 32);
  return v;
}

// ---------------------------------------------------------------------------
// Projection (Q/K): D = W (DKxM) x X^T (Mx16rows) -> tiles [dim x row];
// store f16 row-major out[row][dim] (contiguous 8-half stores).
// grid: 128 blocks x 256 threads; wave handles 16 rows x all 128 dims.
// ---------------------------------------------------------------------------
__global__ void __launch_bounds__(256)
proj_qk_kernel(const float* __restrict__ X, const float* __restrict__ W,
               _Float16* __restrict__ out) {
  const int lane = threadIdx.x & 31;
  const int wv   = threadIdx.x >> 5;
  const int r0   = blockIdx.x * 128 + wv * 16;   // global row (b*S + s)
  const int hi = (lane >> 4) & 1, lr = lane & 15;

  v8f z = {};
  v8f acc[8];
#pragma unroll
  for (int dt = 0; dt < 8; ++dt) acc[dt] = z;

  for (int kk = 0; kk < MDIM; kk += 32) {
    v16h bfrag = load_b_f32t(X, MDIM, r0, kk, lane);    // X rows as B columns
#pragma unroll
    for (int dt = 0; dt < 8; ++dt) {
      v16h afrag = load_a_f32(W, MDIM, dt * 16, kk, lane);
      acc[dt] = WMMA_F16(afrag, bfrag, acc[dt]);
    }
  }
#pragma unroll
  for (int dt = 0; dt < 8; ++dt) {
    _Float16* dst = out + (size_t)(r0 + lr) * DK + dt * 16 + hi * 8;
    store8_f16(dst, acc[dt]);
  }
}

// ---------------------------------------------------------------------------
// Projection (V): D = X (16rows x M) x W^T (M x DK) -> tiles [row x dim];
// store f16 transposed vT[b][dim][s] (contiguous along s).
// ---------------------------------------------------------------------------
__global__ void __launch_bounds__(256)
proj_v_kernel(const float* __restrict__ X, const float* __restrict__ W,
              _Float16* __restrict__ vT) {
  const int lane = threadIdx.x & 31;
  const int wv   = threadIdx.x >> 5;
  const int r0   = blockIdx.x * 128 + wv * 16;
  const int hi = (lane >> 4) & 1, lr = lane & 15;

  v8f z = {};
  v8f acc[8];
#pragma unroll
  for (int dt = 0; dt < 8; ++dt) acc[dt] = z;

  for (int kk = 0; kk < MDIM; kk += 32) {
    v16h afrag = load_a_f32(X, MDIM, r0, kk, lane);
#pragma unroll
    for (int dt = 0; dt < 8; ++dt) {
      v16h bfrag = load_b_f32t(W, MDIM, dt * 16, kk, lane); // W rows = dims
      acc[dt] = WMMA_F16(afrag, bfrag, acc[dt]);
    }
  }
  const int bb = r0 >> 11;          // batch
  const int sb = r0 & (SEQ - 1);    // row within batch
#pragma unroll
  for (int dt = 0; dt < 8; ++dt) {
    _Float16* dst = vT + ((size_t)bb * DK + dt * 16 + lr) * SEQ + sb + hi * 8;
    store8_f16(dst, acc[dt]);
  }
}

// ---------------------------------------------------------------------------
// Flash attention: block = 8 waves = 128 queries of one batch.
// Wave: 16-query tile, key tiles of 32, online softmax, O in 8 f32 fragments.
// ---------------------------------------------------------------------------
__global__ void __launch_bounds__(256)
attn_kernel(const _Float16* __restrict__ qf, const _Float16* __restrict__ kf,
            const _Float16* __restrict__ vT, const int* __restrict__ mask,
            float* __restrict__ out) {
  __shared__ __align__(64) _Float16 pbuf[8 * 16 * 32];   // per-wave P tile

  const int lane = threadIdx.x & 31;
  const int wv   = threadIdx.x >> 5;
  const int b    = blockIdx.x >> 4;
  const int qch  = blockIdx.x & 15;
  const int qb   = qch * 128 + wv * 16;            // query base within batch
  const int hi = (lane >> 4) & 1, lr = lane & 15;

  const _Float16* qbase = qf + (size_t)b * SEQ * DK;
  const _Float16* kbase = kf + (size_t)b * SEQ * DK;
  const _Float16* vbase = vT + (size_t)b * DK * SEQ;
  const int*      mbase = mask + (size_t)b * SEQ * SEQ;
  _Float16*       myp   = pbuf + wv * (16 * 32);

  // Q fragments (16 x 128 in 4 K-chunks), loaded once
  v16h aq[4];
#pragma unroll
  for (int c = 0; c < 4; ++c) aq[c] = load_a_f16(qbase, DK, qb, c * 32, lane);

  v8f z = {};
  v8f o[8];
#pragma unroll
  for (int f = 0; f < 8; ++f) o[f] = z;
  float m_run[8], l_run[8];
#pragma unroll
  for (int i = 0; i < 8; ++i) { m_run[i] = -1e30f; l_run[i] = 0.0f; }

  const float scale = 0.08838834764831845f;   // 1/sqrt(128)

  for (int j = 0; j < SEQ / 32; ++j) {
    const int kb = j * 32;
    // prefetch next key/value tiles
    __builtin_prefetch(kbase + (size_t)(kb + 32) * DK, 0, 1);
    __builtin_prefetch(vbase + (size_t)kb + 32, 0, 1);

    // ---- S = Q K^T (16 x 32), two 16-column fragments -------------------
    v8f s0 = z, s1 = z;
#pragma unroll
    for (int c = 0; c < 4; ++c) {
      v16h bk0 = load_b_f16(kbase, DK, kb,      c * 32, lane);
      v16h bk1 = load_b_f16(kbase, DK, kb + 16, c * 32, lane);
      s0 = WMMA_F16(aq[c], bk0, s0);
      s1 = WMMA_F16(aq[c], bk1, s1);
    }

    // ---- scale + mask ---------------------------------------------------
#pragma unroll
    for (int i = 0; i < 8; ++i) {
      const int qrow = qb + i + 8 * hi;
      const int* mrow = mbase + (size_t)qrow * SEQ + kb;
      s0[i] = (mrow[lr]      != 0) ? s0[i] * scale : -1e9f;
      s1[i] = (mrow[16 + lr] != 0) ? s1[i] * scale : -1e9f;
    }

    // ---- online softmax -------------------------------------------------
    float alpha[8];
    v8f p0, p1;
#pragma unroll
    for (int i = 0; i < 8; ++i) {
      float t = redmax16(fmaxf(s0[i], s1[i]));
      float mn = fmaxf(m_run[i], t);
      alpha[i] = __expf(m_run[i] - mn);
      m_run[i] = mn;
      p0[i] = __expf(s0[i] - mn);
      p1[i] = __expf(s1[i] - mn);
      float rs = redsum16(p0[i] + p1[i]);
      l_run[i] = l_run[i] * alpha[i] + rs;
    }
#pragma unroll
    for (int f = 0; f < 8; ++f)
#pragma unroll
      for (int i = 0; i < 8; ++i) o[f][i] *= alpha[i];

    // ---- P: C-fragment layout -> LDS -> A-fragment layout ---------------
#pragma unroll
    for (int i = 0; i < 8; ++i) {
      myp[(i + 8 * hi) * 32 + lr]      = (_Float16)p0[i];
      myp[(i + 8 * hi) * 32 + 16 + lr] = (_Float16)p1[i];
    }
    __syncthreads();   // all waves iterate in lockstep; orders LDS st->ld
    v16h ap = load_a_f16(myp, 32, 0, 0, lane);

    // ---- O += P V (32 keys x 128 dims) ----------------------------------
#pragma unroll
    for (int f = 0; f < 8; ++f) {
      v16h bv = load_b_f16(vbase, SEQ, f * 16, kb, lane);  // [dim][key]
      o[f] = WMMA_F16(ap, bv, o[f]);
    }
    __syncthreads();
  }

  // ---- normalize + store ------------------------------------------------
  float inv[8];
#pragma unroll
  for (int i = 0; i < 8; ++i) inv[i] = 1.0f / l_run[i];
#pragma unroll
  for (int f = 0; f < 8; ++f)
#pragma unroll
    for (int i = 0; i < 8; ++i) {
      const int qrow = qb + i + 8 * hi;
      out[((size_t)b * SEQ + qrow) * DK + f * 16 + lr] = o[f][i] * inv[i];
    }
}

// ---------------------------------------------------------------------------
extern "C" void kernel_launch(void* const* d_in, const int* in_sizes, int n_in,
                              void* d_out, int out_size, void* d_ws, size_t ws_size,
                              hipStream_t stream) {
  (void)in_sizes; (void)n_in; (void)out_size; (void)ws_size;
  const float* query = (const float*)d_in[0];
  const float* key_  = (const float*)d_in[1];
  const float* value = (const float*)d_in[2];
  const int*   mask  = (const int*)d_in[3];
  const float* Wq    = (const float*)d_in[4];
  const float* Wk    = (const float*)d_in[5];
  const float* Wv    = (const float*)d_in[6];
  float* out = (float*)d_out;

  _Float16* qf = (_Float16*)d_ws;                       // [B*S, 128] f16
  _Float16* kf = qf + (size_t)BATCH * SEQ * DK;         // [B*S, 128] f16
  _Float16* vT = kf + (size_t)BATCH * SEQ * DK;         // [B, 128, S] f16

  dim3 blk(256);
  proj_qk_kernel<<<128, blk, 0, stream>>>(query, Wq, qf);
  proj_qk_kernel<<<128, blk, 0, stream>>>(key_, Wk, kf);
  proj_v_kernel<<<128, blk, 0, stream>>>(value, Wv, vT);
  attn_kernel<<<128, blk, 0, stream>>>(qf, kf, vT, mask, out);
}
